// PRGC_17772574671485
// MI455X (gfx1250) — matrix operand order repaired and checked by reading, not compiled, not run on hardware
//
#include <hip/hip_runtime.h>
#include <hip/hip_bf16.h>

typedef __attribute__((ext_vector_type(16))) _Float16 v16h;
typedef __attribute__((ext_vector_type(8)))  _Float16 v8h;
typedef __attribute__((ext_vector_type(4)))  _Float16 v4h;
typedef __attribute__((ext_vector_type(8)))  float    v8f;

#define Hdim 768
#define Ldim 128
#define Bdim 8
#define BLrows 1024   // B*L
#define MATf  ((size_t)BLrows * Hdim)   // 786432 elements per activation matrix
#define WELEM ((size_t)Hdim * Hdim)     // 589824 elements per weight matrix

// -------------------------------------------------------------------------
// Kernel 0: pooling + stage1 head + per-batch relation bias rows
// -------------------------------------------------------------------------
__global__ __launch_bounds__(256)
void prep_kernel(const float* __restrict__ E, const float* __restrict__ mask,
                 const int* __restrict__ trel,
                 const float* __restrict__ W_hr, const float* __restrict__ b_hr,
                 const float* __restrict__ W_rj, const float* __restrict__ b_rj,
                 const float* __restrict__ rel_emb,
                 const float* __restrict__ W_sh, const float* __restrict__ b_sh,
                 const float* __restrict__ W_oh, const float* __restrict__ b_oh,
                 float* __restrict__ out_stage1,
                 float* __restrict__ rsh, float* __restrict__ roh)
{
    __shared__ float pool[Hdim];
    __shared__ float reb[Hdim];
    __shared__ float rh[Hdim / 2];
    const int b = blockIdx.x, t = threadIdx.x;

    float msum = 0.f;
    for (int l = 0; l < Ldim; ++l) msum += mask[b * Ldim + l];

    for (int h = t; h < Hdim; h += 256) {
        float s = 0.f;
        for (int l = 0; l < Ldim; ++l)
            s += E[((size_t)(b * Ldim + l)) * Hdim + h] * mask[b * Ldim + l];
        pool[h] = s / msum;
    }
    const int r = trel[b];
    for (int h = t; h < Hdim; h += 256) reb[h] = rel_emb[(size_t)r * Hdim + h];
    __syncthreads();

    for (int j = t; j < Hdim / 2; j += 256) {
        float s = b_hr[j];
        for (int h = 0; h < Hdim; ++h) s += pool[h] * W_hr[h * (Hdim / 2) + j];
        rh[j] = fmaxf(s, 0.f);
    }
    __syncthreads();

    for (int j = t; j < 24; j += 256) {
        float s = b_rj[j];
        for (int rr = 0; rr < Hdim / 2; ++rr) s += rh[rr] * W_rj[rr * 24 + j];
        out_stage1[b * 24 + j] = s;
    }

    for (int n = t; n < Hdim; n += 256) {
        float s1 = b_sh[n], s2 = b_oh[n];
        for (int h = 0; h < Hdim; ++h) {
            const float rv = reb[h];
            s1 += rv * W_sh[h * Hdim + n];
            s2 += rv * W_oh[h * Hdim + n];
        }
        rsh[b * Hdim + n] = s1;
        roh[b * Hdim + n] = s2;
    }
}

// -------------------------------------------------------------------------
// Kernel 1a: convert embed (1024x768 f32) -> f16, same layout
// -------------------------------------------------------------------------
__global__ __launch_bounds__(256)
void cvt_embed_kernel(const float* __restrict__ E, _Float16* __restrict__ Ef16)
{
    const size_t i = ((size_t)blockIdx.x * 256 + threadIdx.x) * 4;
    const float4 v = *(const float4*)(E + i);
    v4h o = { (_Float16)v.x, (_Float16)v.y, (_Float16)v.z, (_Float16)v.w };
    *(v4h*)(Ef16 + i) = o;
}

// -------------------------------------------------------------------------
// Kernel 1b: convert + transpose the 4 GEMM weights to f16 [n][k] layout.
// 32x32 LDS tile transpose; coalesced on both sides.
// -------------------------------------------------------------------------
__global__ __launch_bounds__(256)
void wtrans_kernel(const float* __restrict__ w0, const float* __restrict__ w1,
                   const float* __restrict__ w2, const float* __restrict__ w3,
                   _Float16* __restrict__ Wt)
{
    __shared__ float tile[32][33];
    const float* W = (blockIdx.z == 0) ? w0 : (blockIdx.z == 1) ? w1
                   : (blockIdx.z == 2) ? w2 : w3;
    _Float16* D = Wt + (size_t)blockIdx.z * WELEM;
    const int k0 = blockIdx.x * 32, n0 = blockIdx.y * 32;
    const int t = threadIdx.x;
    {
        const int nn = t & 31, kb = t >> 5;
        for (int r = 0; r < 4; ++r) {
            const int kk = kb + r * 8;
            tile[kk][nn] = W[(size_t)(k0 + kk) * Hdim + n0 + nn];
        }
    }
    __syncthreads();
    {
        const int kk = t & 31, nb = t >> 5;
        for (int r = 0; r < 4; ++r) {
            const int nn = nb + r * 8;
            D[(size_t)(n0 + nn) * Hdim + k0 + kk] = (_Float16)tile[kk][nn];
        }
    }
}

// -------------------------------------------------------------------------
// Kernel 2: fused quad-GEMM via WMMA.  Y_z(1024x768) = embed @ W_z(768x768)
// Pure-f16 operands (pre-converted), B pre-transposed -> staging is a raw
// byte copy done with GLOBAL_LOAD_ASYNC_TO_LDS_B128 (ASYNCcnt), double-
// buffered so tile i+1 DMA overlaps tile i WMMA.
// 128x128 block tile, 8 waves (4M x 2N), each wave 2x4 accumulators, BK=32.
// -------------------------------------------------------------------------
#define LDA 40
#define LDB 40
#define KTILES (Hdim / 32)

__device__ __forceinline__
void stage_tile_async(const _Float16* __restrict__ Ef16,
                      const _Float16* __restrict__ Wz,
                      int row0, int col0, int k0,
                      unsigned asBase, unsigned bsBase, int t)
{
    // A tile: 128x32 f16, rows of embed.  B tile: 128(n) x 32(k) of Wt.
    // 512 x 16B chunks each; per thread 2 A + 2 B async instructions.
    for (int rep = 0; rep < 2; ++rep) {
        const int c  = t + rep * 256;
        const int m  = c >> 2;
        const int kk = (c & 3) * 8;
        const _Float16* ga = Ef16 + (size_t)(row0 + m) * Hdim + k0 + kk;
        const unsigned  la = asBase + (unsigned)(m * LDA + kk) * 2u;
        asm volatile("global_load_async_to_lds_b128 %0, %1, off"
                     :: "v"(la), "v"(ga) : "memory");
        const _Float16* gb = Wz + (size_t)(col0 + m) * Hdim + k0 + kk;
        const unsigned  lb = bsBase + (unsigned)(m * LDB + kk) * 2u;
        asm volatile("global_load_async_to_lds_b128 %0, %1, off"
                     :: "v"(lb), "v"(gb) : "memory");
    }
}

__global__ __launch_bounds__(256)
void gemm_wmma(const _Float16* __restrict__ Ef16,
               const _Float16* __restrict__ Wt,
               float* __restrict__ ws)
{
    __shared__ __align__(16) _Float16 As[2][128 * LDA];  // [m][k]
    __shared__ __align__(16) _Float16 Bs[2][128 * LDB];  // [n][k] (pre-transposed)

    const _Float16* Wz = Wt + (size_t)blockIdx.z * WELEM;
    float* Y = ws + (size_t)blockIdx.z * MATf;

    const int t    = threadIdx.x;
    const int wave = t >> 5, lane = t & 31;
    const int wm   = wave >> 1;     // 0..3  (M direction)
    const int wn   = wave & 1;      // 0..1  (N direction)
    const int lr   = lane & 15, half = lane >> 4;
    const int row0 = blockIdx.x * 128;
    const int col0 = blockIdx.y * 128;

    const unsigned asB[2] = { (unsigned)(unsigned long long)&As[0][0],
                              (unsigned)(unsigned long long)&As[1][0] };
    const unsigned bsB[2] = { (unsigned)(unsigned long long)&Bs[0][0],
                              (unsigned)(unsigned long long)&Bs[1][0] };

    const v8f zero = {};
    v8f acc[2][4];
    for (int i = 0; i < 2; ++i)
        for (int j = 0; j < 4; ++j) acc[i][j] = zero;

    // prologue: DMA tile 0 into buffer 0
    stage_tile_async(Ef16, Wz, row0, col0, 0, asB[0], bsB[0], t);

    for (int it = 0; it < KTILES; ++it) {
        const int buf = it & 1;
        if (it + 1 < KTILES) {
            // prefetch next K-tile into the other buffer
            stage_tile_async(Ef16, Wz, row0, col0, (it + 1) * 32,
                             asB[buf ^ 1], bsB[buf ^ 1], t);
            // 4 newest ops belong to the prefetch; in-order retirement means
            // count<=4 <=> current tile's 4 copies have landed in LDS.
            asm volatile("s_wait_asynccnt 0x4" ::: "memory");
        } else {
            asm volatile("s_wait_asynccnt 0x0" ::: "memory");
        }
        __syncthreads();

        // ---- fragments per ISA 7.12.2 layouts ----
        v16h af[2], bf[4];
        for (int i = 0; i < 2; ++i) {
            const _Float16* ap = As[buf] + (wm * 32 + i * 16 + lr) * LDA;
            v8h lo = *(const v8h*)(ap + half * 8);        // K = 8*half .. +7
            v8h hi = *(const v8h*)(ap + 16 + half * 8);   // K = 16+8*half .. +7
            af[i] = __builtin_shufflevector(lo, hi, 0,1,2,3,4,5,6,7,8,9,10,11,12,13,14,15);
        }
        for (int j = 0; j < 4; ++j) {
            const _Float16* bp = Bs[buf] + (wn * 64 + j * 16 + lr) * LDB + half * 8;
            v8h lo = *(const v8h*)(bp);                   // K = 8*half .. +7
            v8h hi = *(const v8h*)(bp + 16);              // K = 16+8*half .. +7
            bf[j] = __builtin_shufflevector(lo, hi, 0,1,2,3,4,5,6,7,8,9,10,11,12,13,14,15);
        }

        for (int i = 0; i < 2; ++i)
            for (int j = 0; j < 4; ++j)
                acc[i][j] = __builtin_amdgcn_wmma_f32_16x16x32_f16(
                    false, af[i], false, bf[j], (short)0, acc[i][j], false, false);
        __syncthreads();   // protect buf from being overwritten by it+2 prefetch
    }

    // ---- epilogue: C/D layout: VGPR v -> row v + 8*half, col lr ----
    for (int i = 0; i < 2; ++i) {
        const int mBase = row0 + wm * 32 + i * 16;
        for (int j = 0; j < 4; ++j) {
            const int n = col0 + wn * 64 + j * 16 + lr;
            for (int v = 0; v < 8; ++v)
                Y[(size_t)(mBase + v + half * 8) * Hdim + n] = acc[i][j][v];
        }
    }
}

// -------------------------------------------------------------------------
// Kernel 3: pred_corres[b,i,j] = b_gc + sum_h relu(A[b,j,h]+C[b,i,h]+b_corr[h])*W_gc[h]
// -------------------------------------------------------------------------
#define CH 256
#define PITCH 257
__global__ __launch_bounds__(256)
void corres_kernel(const float* __restrict__ A, const float* __restrict__ C,
                   const float* __restrict__ b_corr,
                   const float* __restrict__ W_gc, const float* __restrict__ b_gc,
                   float* __restrict__ outP)
{
    __shared__ float aT[16 * PITCH];
    __shared__ float cT[16 * PITCH];
    __shared__ float wg[CH];

    const int jt = blockIdx.x, it = blockIdx.y, b = blockIdx.z;
    const int t = threadIdx.x;
    const int i = t >> 4, j = t & 15;

    float acc = 0.f;
    for (int h0 = 0; h0 < Hdim; h0 += CH) {
        for (int idx = t; idx < 16 * CH; idx += 256) {
            const int rr = idx >> 8, hh = idx & (CH - 1);
            aT[rr * PITCH + hh] =
                A[((size_t)(b * Ldim + jt * 16 + rr)) * Hdim + h0 + hh] + b_corr[h0 + hh];
            cT[rr * PITCH + hh] =
                C[((size_t)(b * Ldim + it * 16 + rr)) * Hdim + h0 + hh];
        }
        if (t < CH) wg[t] = W_gc[h0 + t];
        __syncthreads();

        const float* ap = aT + j * PITCH;
        const float* cp = cT + i * PITCH;
        #pragma unroll 8
        for (int hh = 0; hh < CH; ++hh)
            acc += fmaxf(ap[hh] + cp[hh], 0.f) * wg[hh];
        __syncthreads();
    }
    outP[((size_t)(b * Ldim + it * 16 + i)) * Ldim + jt * 16 + j] = acc + b_gc[0];
}

// -------------------------------------------------------------------------
// Kernel 4: subj/obj heads
// -------------------------------------------------------------------------
__global__ __launch_bounds__(256)
void head_kernel(const float* __restrict__ Hs, const float* __restrict__ Ho,
                 const float* __restrict__ rsh, const float* __restrict__ roh,
                 const float* __restrict__ W_st, const float* __restrict__ b_st,
                 const float* __restrict__ W_ot, const float* __restrict__ b_ot,
                 float* __restrict__ outSubj, float* __restrict__ outObj)
{
    const int g = blockIdx.x * 256 + threadIdx.x;
    const int row = g >> 3, slot = g & 7;
    if (row >= BLrows || slot >= 6) return;
    const int b = row >> 7;
    const bool isS = slot < 3;
    const int tt = isS ? slot : slot - 3;

    const float* Hp = (isS ? Hs : Ho) + (size_t)row * Hdim;
    const float* rp = (isS ? rsh : roh) + (size_t)b * Hdim;
    const float* Wp = isS ? W_st : W_ot;
    float acc = isS ? b_st[tt] : b_ot[tt];
    #pragma unroll 4
    for (int n = 0; n < Hdim; ++n)
        acc += fmaxf(Hp[n] + rp[n], 0.f) * Wp[n * 3 + tt];
    (isS ? outSubj : outObj)[(size_t)row * 3 + tt] = acc;
}

// -------------------------------------------------------------------------
extern "C" void kernel_launch(void* const* d_in, const int* in_sizes, int n_in,
                              void* d_out, int out_size, void* d_ws, size_t ws_size,
                              hipStream_t stream)
{
    (void)in_sizes; (void)n_in; (void)out_size; (void)ws_size;

    const float* E       = (const float*)d_in[0];
    const float* mask    = (const float*)d_in[1];
    const int*   trel    = (const int*)  d_in[2];
    const float* W_hr    = (const float*)d_in[3];
    const float* b_hr    = (const float*)d_in[4];
    const float* W_rj    = (const float*)d_in[5];
    const float* b_rj    = (const float*)d_in[6];
    const float* rel_emb = (const float*)d_in[7];
    const float* W_corr  = (const float*)d_in[8];
    const float* b_corr  = (const float*)d_in[9];
    const float* W_gc    = (const float*)d_in[10];
    const float* b_gc    = (const float*)d_in[11];
    const float* W_sh    = (const float*)d_in[12];
    const float* b_sh    = (const float*)d_in[13];
    const float* W_st    = (const float*)d_in[14];
    const float* b_st    = (const float*)d_in[15];
    const float* W_oh    = (const float*)d_in[16];
    const float* b_oh    = (const float*)d_in[17];
    const float* W_ot    = (const float*)d_in[18];
    const float* b_ot    = (const float*)d_in[19];

    // output tuple, flat: stage1(8x24) | subj(8x128x3) | obj(8x128x3) | pred(8x128x128)
    float* out    = (float*)d_out;
    float* stage1 = out;
    float* subj   = out + 192;
    float* obj    = out + 192 + 3072;
    float* pred   = out + 192 + 3072 + 3072;

    // workspace (f32): A | C | Hs | Ho | rsh | roh, then (f16): Ef16 | Wt[4]
    float* ws   = (float*)d_ws;
    float* Amat = ws;
    float* Cmat = ws + MATf;
    float* Hsm  = ws + 2 * MATf;
    float* Hom  = ws + 3 * MATf;
    float* rsh  = ws + 4 * MATf;
    float* roh  = rsh + Bdim * Hdim;
    _Float16* Ef16 = (_Float16*)(roh + Bdim * Hdim);
    _Float16* Wt   = Ef16 + MATf;

    prep_kernel<<<Bdim, 256, 0, stream>>>(E, mask, trel, W_hr, b_hr, W_rj, b_rj,
                                          rel_emb, W_sh, b_sh, W_oh, b_oh,
                                          stage1, rsh, roh);

    cvt_embed_kernel<<<MATf / 1024, 256, 0, stream>>>(E, Ef16);

    dim3 gt(Hdim / 32, Hdim / 32, 4);
    wtrans_kernel<<<gt, 256, 0, stream>>>(W_corr,                 // z=0 -> A
                                          W_corr + Hdim * Hdim,   // z=1 -> C
                                          W_sh   + Hdim * Hdim,   // z=2 -> Hs
                                          W_oh   + Hdim * Hdim,   // z=3 -> Ho
                                          Wt);

    dim3 g1(BLrows / 128, Hdim / 128, 4);
    gemm_wmma<<<g1, 256, 0, stream>>>(Ef16, Wt, ws);

    dim3 g2(Ldim / 16, Ldim / 16, Bdim);
    corres_kernel<<<g2, 256, 0, stream>>>(Amat, Cmat, b_corr, W_gc, b_gc, pred);

    head_kernel<<<(BLrows * 8) / 256, 256, 0, stream>>>(Hsm, Hom, rsh, roh,
                                                        W_st, b_st, W_ot, b_ot,
                                                        subj, obj);
}
